// SwinBlock3D_3435973836950
// MI455X (gfx1250) — matrix operand order repaired and checked by reading, not compiled, not run on hardware
//
#include <hip/hip_runtime.h>
#include <hip/hip_bf16.h>

// ---------------------------------------------------------------------------
// Swin-3D block for MI455X (gfx1250, wave32, WMMA).
// All GEMMs: v_wmma_f32_16x16x32_f16, f32 accumulate.
// B operands staged TRANSPOSED in LDS so every fragment load is contiguous
// (2x ds_load_b128 per fragment). A panels staged via async global->LDS DMA.
// Token dim padded 196 -> 224 (14 M-tiles, 7 K-chunks of 32).
// ---------------------------------------------------------------------------

typedef __attribute__((ext_vector_type(16))) _Float16 v16h;
typedef __attribute__((ext_vector_type(8)))  float    v8f;
typedef _Float16 half_t;

#define LT   196
#define LP   224
#define CEMB 128
#define PDIM 256
#define NHEAD 8
#define HDIM 32
#define HIDW 384
#define NWIN 128
#define NBW  256
#define D0 8
#define H0 56
#define W0 56
#define SD 2
#define SH 3
#define SW 3

// ---------------- workspace layout (bytes), regions time-aliased ----------
#define OFF_BIAS ((size_t)0)
#define OFF_MASK ((size_t)1  << 20)
#define OFF_QW   ((size_t)16 << 20)
#define OFF_KW   ((size_t)32 << 20)
#define OFF_VW   ((size_t)48 << 20)
#define OFF_QP   ((size_t)64 << 20)
#define OFF_KP   ((size_t)96 << 20)
#define OFF_VP   ((size_t)128 << 20)
#define OFF_CTX  ((size_t)160 << 20)
#define OFF_A16  OFF_QW
#define OFF_A32  OFF_KW
#define OFF_HM   OFF_QP
#define OFF_MO   OFF_VP

// ---------------- CDNA5 async global->LDS staging --------------------------
// ISA 15.18.3 op 98: GLOBAL_LOAD_ASYNC_TO_LDS_B128 (ASYNCcnt tracked).
__device__ inline void async_ld_b128(void* lds_ptr, const void* gptr) {
  uint32_t off = (uint32_t)(uintptr_t)lds_ptr;   // LDS aperture: addr[31:0]
  asm volatile("global_load_async_to_lds_b128 %0, %1, off"
               :: "v"(off), "v"(gptr) : "memory");
}
__device__ inline void async_wait0() {
  asm volatile("s_wait_asynccnt 0" ::: "memory");
}

// ---------------- WMMA fragment helpers (contiguous k order: k = 16*(lane/16) + s) ---------------

// A fragment: 16x32 f16, A row-major in LDS with stride lda (halfs).
__device__ inline v16h load_a_frag(const half_t* A, int lda, int lane) {
  int m    = lane & 15;
  int ksel = (lane >> 4) << 4;
  const half_t* row = A + m * lda;
  v16h a;
#pragma unroll
  for (int j = 0; j < 4; ++j) {
    a[4*j]     = row[ksel + 4*j];
    a[4*j+1]   = row[ksel + 4*j + 1];
    a[4*j+2]   = row[ksel + 4*j + 2];
    a[4*j+3]   = row[ksel + 4*j + 3];
  }
  return a;
}

// B fragment from TRANSPOSED panel BT[N][K]: element (k,n) = BT[n][k].
// 16 contiguous halfs per lane -> ds_load_b128 pairs.
__device__ inline v16h load_bT_frag(const half_t* BTrow, int lane) {
  int n  = lane & 15;
  int kb = (lane >> 4) << 4;
  const half_t* row = BTrow + kb;
  (void)n;
  v16h b;
#pragma unroll
  for (int j = 0; j < 16; ++j) b[j] = row[j];
  return b;
}

// D += A(16 x kTot) * B(kTot x 16); BT is [N][K] transposed, stride ldt.
__device__ inline v8f wmma_tile_t(const half_t* A, int lda,
                                  const half_t* BT, int ldt,
                                  int kTot, int lane, v8f acc) {
  int n  = lane & 15;
  int kb = (lane >> 4) << 4;
  const half_t* brow = BT + (size_t)n * ldt + kb;
  for (int k0 = 0; k0 < kTot; k0 += 32) {
    v16h a = load_a_frag(A + k0, lda, lane);
    v16h b;
#pragma unroll
    for (int j = 0; j < 16; ++j) b[j] = brow[k0 + j];
    acc = __builtin_amdgcn_wmma_f32_16x16x32_f16(false, a, false, b,
                                                 (short)0, acc, false, false);
  }
  return acc;
}

// ---------------- k0: bias/mask tables -> padded f16 -----------------------
__global__ void k0_prep(const float* __restrict__ rel_table,
                        const int*   __restrict__ rel_idx,
                        const float* __restrict__ attn_mask,
                        half_t* __restrict__ biasF,
                        half_t* __restrict__ maskF) {
  size_t gid = (size_t)blockIdx.x * blockDim.x + threadIdx.x;
  size_t stride = (size_t)gridDim.x * blockDim.x;
  size_t nb = (size_t)NHEAD * LP * LP;
  for (size_t i = gid; i < nb; i += stride) {
    int k = (int)(i % LP); size_t r = i / LP; int q = (int)(r % LP); int h = (int)(r / LP);
    float v = 0.f;
    if (q < LT && k < LT) v = rel_table[rel_idx[q * LT + k] * NHEAD + h];
    biasF[i] = (half_t)v;
  }
  size_t nm = (size_t)NWIN * LP * LP;
  for (size_t i = gid; i < nm; i += stride) {
    int k = (int)(i % LP); size_t r = i / LP; int q = (int)(r % LP); int w = (int)(r / LP);
    float v = 0.f;
    if (q < LT && k < LT) v = attn_mask[((size_t)w * LT + q) * LT + k];
    maskF[i] = (half_t)v;
  }
}

// ---------------- k1: roll + window partition, fp32 -> f16 -----------------
__global__ void k1_gather(const float* __restrict__ q,
                          const float* __restrict__ k,
                          const float* __restrict__ v,
                          half_t* __restrict__ qw,
                          half_t* __restrict__ kw,
                          half_t* __restrict__ vw) {
  size_t gid = (size_t)blockIdx.x * blockDim.x + threadIdx.x;
  size_t tot = (size_t)NBW * LP * CEMB;
  if (gid >= tot) return;
  int c = (int)(gid & (CEMB - 1));
  int t = (int)((gid >> 7) % LP);
  int n = (int)(gid / ((size_t)LP * CEMB));
  if (t >= LT) { qw[gid] = (half_t)0.f; kw[gid] = (half_t)0.f; vw[gid] = (half_t)0.f; return; }
  int b  = n >> 7, w = n & (NWIN - 1);
  int wd = w >> 6, wh = (w >> 3) & 7, ww = w & 7;
  int td = t / 49, th = (t / 7) % 7, tw = t % 7;
  int d  = (wd * 4 + td + SD) & 7;
  int hh = (wh * 7 + th + SH) % H0;
  int wi = (ww * 7 + tw + SW) % W0;
  size_t src = ((((size_t)b * D0 + d) * H0 + hh) * W0 + wi) * CEMB + c;
  qw[gid] = (half_t)q[src];
  kw[gid] = (half_t)k[src];
  vw[gid] = (half_t)v[src];
}

// ---------------- k2: per-window QKV projection GEMM -----------------------
__global__ void k2_qkv(const half_t* __restrict__ Xw,
                       const float*  __restrict__ W,      // [CEMB][PDIM]
                       const float*  __restrict__ bias,
                       half_t* __restrict__ Yp) {
  extern __shared__ char smem[];
  half_t* sA  = (half_t*)smem;                    // LP*CEMB
  half_t* sBT = sA + (size_t)LP * CEMB;           // [PDIM][CEMB] transposed
  int n = blockIdx.x, tid = threadIdx.x;
  const char* gA = (const char*)(Xw + (size_t)n * LP * CEMB);
  for (int i = tid; i < LP * CEMB / 8; i += 256)
    async_ld_b128((char*)sA + i * 16, gA + i * 16);
  for (int i = tid; i < CEMB * PDIM; i += 256) {
    int kk = i >> 8, nn = i & (PDIM - 1);
    sBT[(size_t)nn * CEMB + kk] = (half_t)W[i];
  }
  async_wait0();
  __syncthreads();
  int lane = tid & 31, wave = tid >> 5;
  for (int t = wave; t < 14 * (PDIM / 16); t += 8) {
    int mt = t % 14, nt = t / 14;
    float bn = bias[nt * 16 + (lane & 15)];
    v8f acc;
#pragma unroll
    for (int r = 0; r < 8; ++r) acc[r] = bn;
    acc = wmma_tile_t(sA + (size_t)mt * 16 * CEMB, CEMB,
                      sBT + (size_t)(nt * 16) * CEMB, CEMB, CEMB, lane, acc);
    half_t* out = Yp + ((size_t)n * LP + mt * 16) * PDIM + nt * 16;
    int nn = lane & 15, moff = (lane >> 4) * 8;
#pragma unroll
    for (int r = 0; r < 8; ++r) out[(size_t)(moff + r) * PDIM + nn] = (half_t)acc[r];
  }
}

// ---------------- k3: attention, one block per (window, head) --------------
__global__ void k3_attn(const half_t* __restrict__ Qp,
                        const half_t* __restrict__ Kp,
                        const half_t* __restrict__ Vp,
                        const half_t* __restrict__ biasF,
                        const half_t* __restrict__ maskF,
                        half_t* __restrict__ Ctx) {
  extern __shared__ char smem[];
  half_t* sQ  = (half_t*)smem;                 // [LP][HDIM]
  half_t* sK  = sQ + (size_t)LP * HDIM;        // [LP][HDIM]
  half_t* sVT = sK + (size_t)LP * HDIM;        // [HDIM][LP] transposed
  half_t* sS  = sVT + (size_t)HDIM * LP;       // [LP][LP]
  int n = blockIdx.x >> 3, h = blockIdx.x & 7;
  int w = n & (NWIN - 1);
  int tid = threadIdx.x;
  // Q/K head-slabs: 64B rows -> async DMA, 4 x b128 per row.
  for (int i = tid; i < LP * 4; i += 256) {
    int t = i >> 2, c = i & 3;
    size_t gbase = ((size_t)n * LP + t) * PDIM + h * HDIM;
    async_ld_b128((char*)sQ + t * 64 + c * 16, (const char*)(Qp + gbase) + c * 16);
    async_ld_b128((char*)sK + t * 64 + c * 16, (const char*)(Kp + gbase) + c * 16);
  }
  // V slab transposed through VGPRs (coalesced reads, strided LDS writes).
  for (int i = tid; i < LP * HDIM; i += 256) {
    int t = i >> 5, d = i & 31;
    sVT[(size_t)d * LP + t] = Vp[((size_t)n * LP + t) * PDIM + h * HDIM + d];
  }
  async_wait0();
  __syncthreads();
  int lane = tid & 31, wave = tid >> 5;
  const float scale = 0.17677669529663687f;
  // logits: S = Q @ K^T  (contraction = 32, B frag from sK rows -> contiguous)
  for (int t = wave; t < 14 * 14; t += 8) {
    int mt = t % 14, nt = t / 14;
    v16h a = load_a_frag(sQ + (size_t)mt * 16 * HDIM, HDIM, lane);
    v16h b = load_bT_frag(sK + (size_t)(nt * 16 + (lane & 15)) * HDIM, lane);
    v8f acc = {};
    acc = __builtin_amdgcn_wmma_f32_16x16x32_f16(false, a, false, b,
                                                 (short)0, acc, false, false);
    int nn = lane & 15, moff = (lane >> 4) * 8;
    int kcol = nt * 16 + nn;
#pragma unroll
    for (int r = 0; r < 8; ++r) {
      int qrow = mt * 16 + moff + r;
      float val = acc[r] * scale;
      val += (float)biasF[((size_t)h * LP + qrow) * LP + kcol];
      val += (float)maskF[((size_t)w * LP + qrow) * LP + kcol];
      if (kcol >= LT) val = -10000.0f;
      sS[(size_t)qrow * LP + kcol] = (half_t)val;
    }
  }
  __syncthreads();
  // row softmax
  if (tid < LP) {
    int qrow = tid;
    half_t* row = sS + (size_t)qrow * LP;
    if (qrow >= LT) {
      for (int c = 0; c < LP; ++c) row[c] = (half_t)0.f;
    } else {
      float mx = -1e30f;
      for (int c = 0; c < LT; ++c) mx = fmaxf(mx, (float)row[c]);
      float sum = 0.f;
      for (int c = 0; c < LT; ++c) { float e = __expf((float)row[c] - mx); sum += e; row[c] = (half_t)e; }
      float inv = 1.0f / sum;
      for (int c = 0; c < LT; ++c) row[c] = (half_t)((float)row[c] * inv);
      for (int c = LT; c < LP; ++c) row[c] = (half_t)0.f;
    }
  }
  __syncthreads();
  // ctx = P @ V  (M=LP, N=32, K=LP) using transposed V panel
  for (int t = wave; t < 14 * 2; t += 8) {
    int mt = t % 14, nt = t / 14;
    v8f acc = {};
    acc = wmma_tile_t(sS + (size_t)mt * 16 * LP, LP,
                      sVT + (size_t)(nt * 16) * LP, LP, LP, lane, acc);
    half_t* out = Ctx + ((size_t)n * LP + mt * 16) * PDIM + h * HDIM + nt * 16;
    int nn = lane & 15, moff = (lane >> 4) * 8;
#pragma unroll
    for (int r = 0; r < 8; ++r) out[(size_t)(moff + r) * PDIM + nn] = (half_t)acc[r];
  }
}

// ---------------- k4: out-proj + LN1 + residual ----------------------------
__global__ void k4_proj(const half_t* __restrict__ Ctx,
                        const float* __restrict__ Wo, const float* __restrict__ bo,
                        const float* __restrict__ qg,
                        const float* __restrict__ g1, const float* __restrict__ bl1,
                        float*  __restrict__ att32,
                        half_t* __restrict__ att16) {
  extern __shared__ char smem[];
  half_t* sA  = (half_t*)smem;                        // LP*PDIM
  half_t* sBT = sA + (size_t)LP * PDIM;               // [CEMB][PDIM] transposed
  float*  sR  = (float*)(sBT + (size_t)PDIM * CEMB);  // LP*CEMB f32
  int n = blockIdx.x, tid = threadIdx.x;
  const char* gA = (const char*)(Ctx + (size_t)n * LP * PDIM);
  for (int i = tid; i < LP * PDIM / 8; i += 256)
    async_ld_b128((char*)sA + i * 16, gA + i * 16);
  for (int i = tid; i < PDIM * CEMB; i += 256) {
    int kk = i >> 7, nn = i & (CEMB - 1);              // Wo[k][n]
    sBT[(size_t)nn * PDIM + kk] = (half_t)Wo[i];
  }
  async_wait0();
  __syncthreads();
  int lane = tid & 31, wave = tid >> 5;
  for (int t = wave; t < 14 * 8; t += 8) {
    int mt = t % 14, nt = t / 14;
    float bn = bo[nt * 16 + (lane & 15)];
    v8f acc;
#pragma unroll
    for (int r = 0; r < 8; ++r) acc[r] = bn;
    acc = wmma_tile_t(sA + (size_t)mt * 16 * PDIM, PDIM,
                      sBT + (size_t)(nt * 16) * PDIM, PDIM, PDIM, lane, acc);
    int nn = lane & 15, moff = (lane >> 4) * 8;
#pragma unroll
    for (int r = 0; r < 8; ++r)
      sR[(size_t)(mt * 16 + moff + r) * CEMB + nt * 16 + nn] = acc[r];
  }
  __syncthreads();
  if (tid < LP) {
    int r = tid;
    if (r >= LT) {
      for (int c = 0; c < CEMB; ++c) att16[((size_t)n * LP + r) * CEMB + c] = (half_t)0.f;
    } else {
      const float* xr = sR + (size_t)r * CEMB;
      float mu = 0.f;
      for (int c = 0; c < CEMB; ++c) mu += xr[c];
      mu *= (1.0f / CEMB);
      float var = 0.f;
      for (int c = 0; c < CEMB; ++c) { float d = xr[c] - mu; var += d * d; }
      float rs = rsqrtf(var * (1.0f / CEMB) + 1e-5f);
      int b  = n >> 7, w = n & (NWIN - 1);
      int wd = w >> 6, wh = (w >> 3) & 7, ww = w & 7;
      int td = r / 49, th = (r / 7) % 7, tw = r % 7;
      int d  = (wd * 4 + td + SD) & 7;
      int hh = (wh * 7 + th + SH) % H0;
      int wi = (ww * 7 + tw + SW) % W0;
      const float* qrow = qg + ((((size_t)b * D0 + d) * H0 + hh) * W0 + wi) * CEMB;
      for (int c = 0; c < CEMB; ++c) {
        float o = (xr[c] - mu) * rs * g1[c] + bl1[c] + qrow[c];
        att32[((size_t)n * LP + r) * CEMB + c] = o;
        att16[((size_t)n * LP + r) * CEMB + c] = (half_t)o;
      }
    }
  }
}

// ---------------- k5: MLP fc1 + exact GELU ---------------------------------
__global__ void k5_fc1(const half_t* __restrict__ A16,
                       const float* __restrict__ W1, const float* __restrict__ b1,
                       half_t* __restrict__ Hm) {
  extern __shared__ char smem[];
  half_t* sA  = (half_t*)smem;                 // LP*CEMB
  half_t* sBT = sA + (size_t)LP * CEMB;        // [HIDW][CEMB] transposed
  int n = blockIdx.x, tid = threadIdx.x;
  const char* gA = (const char*)(A16 + (size_t)n * LP * CEMB);
  for (int i = tid; i < LP * CEMB / 8; i += 256)
    async_ld_b128((char*)sA + i * 16, gA + i * 16);
  for (int i = tid; i < CEMB * HIDW; i += 256) {
    int kk = i / HIDW, nn = i % HIDW;          // W1[k][n]
    sBT[(size_t)nn * CEMB + kk] = (half_t)W1[i];
  }
  async_wait0();
  __syncthreads();
  int lane = tid & 31, wave = tid >> 5;
  for (int t = wave; t < 14 * (HIDW / 16); t += 8) {
    int mt = t % 14, nt = t / 14;
    float bn = b1[nt * 16 + (lane & 15)];
    v8f acc;
#pragma unroll
    for (int r = 0; r < 8; ++r) acc[r] = bn;
    acc = wmma_tile_t(sA + (size_t)mt * 16 * CEMB, CEMB,
                      sBT + (size_t)(nt * 16) * CEMB, CEMB, CEMB, lane, acc);
    half_t* out = Hm + ((size_t)n * LP + mt * 16) * HIDW + nt * 16;
    int nn = lane & 15, moff = (lane >> 4) * 8;
#pragma unroll
    for (int r = 0; r < 8; ++r) {
      float x = acc[r];
      float g = 0.5f * x * (1.0f + erff(x * 0.70710678118654752f));
      out[(size_t)(moff + r) * HIDW + nn] = (half_t)g;
    }
  }
}

// ---------------- k6: MLP fc2 ----------------------------------------------
__global__ void k6_fc2(const half_t* __restrict__ Hm,
                       const float* __restrict__ W2, const float* __restrict__ b2,
                       float* __restrict__ MO) {
  extern __shared__ char smem[];
  half_t* sA  = (half_t*)smem;                 // LP*HIDW
  half_t* sBT = sA + (size_t)LP * HIDW;        // [CEMB][HIDW] transposed
  int n = blockIdx.x, tid = threadIdx.x;
  const char* gA = (const char*)(Hm + (size_t)n * LP * HIDW);
  for (int i = tid; i < LP * HIDW / 8; i += 256)
    async_ld_b128((char*)sA + i * 16, gA + i * 16);
  for (int i = tid; i < HIDW * CEMB; i += 256) {
    int kk = i >> 7, nn = i & (CEMB - 1);      // W2[k][n]
    sBT[(size_t)nn * HIDW + kk] = (half_t)W2[i];
  }
  async_wait0();
  __syncthreads();
  int lane = tid & 31, wave = tid >> 5;
  for (int t = wave; t < 14 * 8; t += 8) {
    int mt = t % 14, nt = t / 14;
    float bn = b2[nt * 16 + (lane & 15)];
    v8f acc;
#pragma unroll
    for (int r = 0; r < 8; ++r) acc[r] = bn;
    acc = wmma_tile_t(sA + (size_t)mt * 16 * HIDW, HIDW,
                      sBT + (size_t)(nt * 16) * HIDW, HIDW, HIDW, lane, acc);
    float* out = MO + ((size_t)n * LP + mt * 16) * CEMB + nt * 16;
    int nn = lane & 15, moff = (lane >> 4) * 8;
#pragma unroll
    for (int r = 0; r < 8; ++r) out[(size_t)(moff + r) * CEMB + nn] = acc[r];
  }
}

// ---------------- k7: LN2 + residual + reverse + unroll scatter ------------
__global__ void k7_final(const float* __restrict__ MO,
                         const float* __restrict__ att32,
                         const float* __restrict__ g2, const float* __restrict__ bl2,
                         float* __restrict__ out) {
  int row = blockIdx.x * 8 + (threadIdx.x >> 5);
  int lane = threadIdx.x & 31;
  if (row >= 2 * D0 * H0 * W0) return;
  int wv = row % W0; int r1 = row / W0;
  int hv = r1 % H0;  r1 /= H0;
  int dv = r1 % D0;  int b = r1 / D0;
  int d2 = (dv + D0 - SD) % D0;
  int h2 = (hv + H0 - SH) % H0;
  int w2 = (wv + W0 - SW) % W0;
  int wd = d2 / 4, wh = h2 / 7, ww = w2 / 7;
  int t  = (d2 % 4) * 49 + (h2 % 7) * 7 + (w2 % 7);
  int n  = b * NWIN + wd * 64 + wh * 8 + ww;
  const float* xr = MO    + ((size_t)n * LP + t) * CEMB;
  const float* ar = att32 + ((size_t)n * LP + t) * CEMB;
  float x[4];
  float s = 0.f;
#pragma unroll
  for (int i = 0; i < 4; ++i) { x[i] = xr[lane * 4 + i]; s += x[i]; }
#pragma unroll
  for (int off = 16; off; off >>= 1) s += __shfl_xor(s, off, 32);
  float mu = s * (1.0f / CEMB);
  float v = 0.f;
#pragma unroll
  for (int i = 0; i < 4; ++i) { float d = x[i] - mu; v += d * d; }
#pragma unroll
  for (int off = 16; off; off >>= 1) v += __shfl_xor(v, off, 32);
  float rs = rsqrtf(v * (1.0f / CEMB) + 1e-5f);
  float* orow = out + ((((size_t)b * D0 + dv) * H0 + hv) * W0 + wv) * CEMB;
#pragma unroll
  for (int i = 0; i < 4; ++i) {
    int c = lane * 4 + i;
    orow[c] = (x[i] - mu) * rs * g2[c] + bl2[c] + ar[c];
  }
}

// ---------------------------------------------------------------------------
extern "C" void kernel_launch(void* const* d_in, const int* in_sizes, int n_in,
                              void* d_out, int out_size, void* d_ws, size_t ws_size,
                              hipStream_t stream) {
  (void)in_sizes; (void)n_in; (void)out_size; (void)ws_size;
  const float* q   = (const float*)d_in[0];
  const float* k   = (const float*)d_in[1];
  const float* v   = (const float*)d_in[2];
  const float* Wq  = (const float*)d_in[3];
  const float* bq  = (const float*)d_in[4];
  const float* Wk  = (const float*)d_in[5];
  const float* bk  = (const float*)d_in[6];
  const float* Wv  = (const float*)d_in[7];
  const float* bv  = (const float*)d_in[8];
  const float* Wo  = (const float*)d_in[9];
  const float* bo  = (const float*)d_in[10];
  const float* rt  = (const float*)d_in[11];
  const float* g1  = (const float*)d_in[12];
  const float* bl1 = (const float*)d_in[13];
  const float* W1  = (const float*)d_in[14];
  const float* b1  = (const float*)d_in[15];
  const float* W2  = (const float*)d_in[16];
  const float* b2  = (const float*)d_in[17];
  const float* g2  = (const float*)d_in[18];
  const float* bl2 = (const float*)d_in[19];
  const int*   ri  = (const int*)d_in[20];
  const float* am  = (const float*)d_in[21];

  char* ws = (char*)d_ws;
  half_t* biasF = (half_t*)(ws + OFF_BIAS);
  half_t* maskF = (half_t*)(ws + OFF_MASK);
  half_t* qw    = (half_t*)(ws + OFF_QW);
  half_t* kw    = (half_t*)(ws + OFF_KW);
  half_t* vw    = (half_t*)(ws + OFF_VW);
  half_t* Qp    = (half_t*)(ws + OFF_QP);
  half_t* Kp    = (half_t*)(ws + OFF_KP);
  half_t* Vp    = (half_t*)(ws + OFF_VP);
  half_t* Ctx   = (half_t*)(ws + OFF_CTX);
  half_t* a16   = (half_t*)(ws + OFF_A16);
  float*  a32   = (float*)(ws + OFF_A32);
  half_t* Hm    = (half_t*)(ws + OFF_HM);
  float*  MO    = (float*)(ws + OFF_MO);

  k0_prep<<<8192, 256, 0, stream>>>(rt, ri, am, biasF, maskF);

  size_t tot1 = (size_t)NBW * LP * CEMB;
  k1_gather<<<(unsigned)((tot1 + 255) / 256), 256, 0, stream>>>(q, k, v, qw, kw, vw);

  size_t sh2 = ((size_t)LP * CEMB + (size_t)CEMB * PDIM) * 2;
  k2_qkv<<<NBW, 256, sh2, stream>>>(qw, Wq, bq, Qp);
  k2_qkv<<<NBW, 256, sh2, stream>>>(kw, Wk, bk, Kp);
  k2_qkv<<<NBW, 256, sh2, stream>>>(vw, Wv, bv, Vp);

  size_t sh3 = ((size_t)2 * LP * HDIM + (size_t)HDIM * LP + (size_t)LP * LP) * 2;
  k3_attn<<<NBW * NHEAD, 256, sh3, stream>>>(Qp, Kp, Vp, biasF, maskF, Ctx);

  size_t sh4 = ((size_t)LP * PDIM + (size_t)PDIM * CEMB) * 2
             + (size_t)LP * CEMB * 4;
  k4_proj<<<NBW, 256, sh4, stream>>>(Ctx, Wo, bo, q, g1, bl1, a32, a16);

  size_t sh5 = ((size_t)LP * CEMB + (size_t)CEMB * HIDW) * 2;
  k5_fc1<<<NBW, 256, sh5, stream>>>(a16, W1, b1, Hm);

  size_t sh6 = ((size_t)LP * HIDW + (size_t)HIDW * CEMB) * 2;
  k6_fc2<<<NBW, 256, sh6, stream>>>(Hm, W2, b2, MO);

  k7_final<<<6272, 256, 0, stream>>>(MO, a32, g2, bl2, (float*)d_out);
}
